// MultiHeadAttention_42013370089935
// MI455X (gfx1250) — matrix operand order repaired and checked by reading, not compiled, not run on hardware
//
#include <hip/hip_runtime.h>
#include <hip/hip_bf16.h>
#include <cstdint>

#define DMODEL 1024
#define SEQ    2048
#define NHEAD  16
#define DHEAD  64

typedef __attribute__((ext_vector_type(16))) __bf16 v16bf;
typedef __attribute__((ext_vector_type(8)))  __bf16 v8bf;
typedef __attribute__((ext_vector_type(8)))  float  v8f;
typedef __attribute__((ext_vector_type(4)))  int    v4i;

// Pointer types matching the async-load builtin signature:
//   (v4i addrspace(1)* global_src, v4i addrspace(3)* lds_dst, imm offset, imm cpol)
typedef __attribute__((address_space(1))) v4i* g_v4i_ptr;
typedef __attribute__((address_space(3))) v4i* l_v4i_ptr;

struct bfpair { v8bf lo; v8bf hi; };

#if defined(__has_builtin)
#if __has_builtin(__builtin_amdgcn_global_load_async_to_lds_b128) && \
    __has_builtin(__builtin_amdgcn_s_wait_asynccnt)
#define HAS_ASYNC_LDS 1
#endif
#endif
#ifndef HAS_ASYNC_LDS
#define HAS_ASYNC_LDS 0
#endif

// Build a 16-element bf16 fragment from two 16-byte (8 x bf16) chunks.
// Elements 0..7 <- p0[0..7], elements 8..15 <- p1[0..7]; element pair (2i,2i+1)
// occupies VGPR i, matching the ISA A/B fragment pair packing.
static __device__ __forceinline__ v16bf frag16(const __bf16* p0, const __bf16* p1) {
  bfpair t;
  t.lo = *(const v8bf*)p0;
  t.hi = *(const v8bf*)p1;
  return __builtin_bit_cast(v16bf, t);
}

// 16-byte global -> LDS copy: async DMA (ASYNCcnt) when available, else bounce.
static __device__ __forceinline__ void copy16_g2l(const __bf16* g, __bf16* l) {
#if HAS_ASYNC_LDS
  __builtin_amdgcn_global_load_async_to_lds_b128((g_v4i_ptr)g, (l_v4i_ptr)l, 0, 0);
#else
  *(v8bf*)l = *(const v8bf*)g;
#endif
}
static __device__ __forceinline__ void wait_async_done() {
#if HAS_ASYNC_LDS
  __builtin_amdgcn_s_wait_asynccnt(0);
#endif
}

static __device__ __forceinline__ float red16_sum(float v) {
#pragma unroll
  for (int m = 1; m < 16; m <<= 1) v += __shfl_xor(v, m, 32);
  return v;
}
static __device__ __forceinline__ float red16_max(float v) {
#pragma unroll
  for (int m = 1; m < 16; m <<= 1) v = fmaxf(v, __shfl_xor(v, m, 32));
  return v;
}

// ---------------------------------------------------------------------------
// Pre-pass: W [k][n] fp32 -> Wt [n][k] bf16  (1024x1024)
// ---------------------------------------------------------------------------
__global__ __launch_bounds__(256) void k_wconv(const float* __restrict__ W,
                                               __bf16* __restrict__ Wt) {
  const int idx = blockIdx.x * 256 + threadIdx.x;
  const int k = idx >> 10, n = idx & 1023;
  Wt[((size_t)n << 10) | (unsigned)k] = (__bf16)W[idx];
}

// ---------------------------------------------------------------------------
// Fused Dense (no bias) + LayerNorm.  X [8192,1024] f32, Wt [n][k] bf16.
// Block: 16 output rows x 1024 cols, 8 waves (each wave 16x128 via 8 WMMA tiles).
// Output bf16 in split-head layout; transpose_out=1 stores [bh][d][s] (for V).
// ---------------------------------------------------------------------------
__global__ __launch_bounds__(256) void k_gemm_ln(
    const float* __restrict__ X, const __bf16* __restrict__ Wt,
    const float* __restrict__ gain, const float* __restrict__ bias,
    __bf16* __restrict__ out, int transpose_out)
{
  __shared__ __bf16 At[16][1032];     // padded row stride: conflict-free b128 reads
  __shared__ float  wstats[8][16][2];
  __shared__ float  fstat[16][2];

  const int tid  = threadIdx.x;
  const int w    = tid >> 5;
  const int lane = tid & 31;
  const int hh   = lane >> 4;
  const int m    = lane & 15;
  const int rowbase = blockIdx.x * 16;

  // Stage 16 rows of X into LDS as bf16.
  const float2* X2 = (const float2*)X;
#pragma unroll 4
  for (int p = tid; p < 16 * 512; p += 256) {
    const int r = p >> 9, cp = p & 511;
    float2 v = X2[(size_t)(rowbase + r) * 512 + cp];
    At[r][2 * cp]     = (__bf16)v.x;
    At[r][2 * cp + 1] = (__bf16)v.y;
  }
  __syncthreads();

  v8f acc[8] = {};
  for (int kk = 0; kk < 32; ++kk) {
    const int kb = kk * 32;
    // A fragment (16x32): lane row m, contiguous k = kb + 16*hh + s, same order as B.
    v16bf afrag = frag16(&At[m][kb + hh * 16], &At[m][kb + hh * 16 + 8]);
#pragma unroll
    for (int nt = 0; nt < 8; ++nt) {
      const __bf16* wp = Wt + (size_t)(w * 128 + nt * 16 + m) * 1024 + kb + hh * 16;
      __builtin_prefetch(wp + 32, 0, 1);            // next K-chunk (global_prefetch_b8)
      v16bf bfrag = frag16(wp, wp + 8);             // B fragment (32x16)
      acc[nt] = __builtin_amdgcn_wmma_f32_16x16x32_bf16(
          false, afrag, false, bfrag, (short)0, acc[nt], false, false);
    }
  }

  // LayerNorm stats straight from C-layout accumulators.
  float psum[8] = {}, psq[8] = {};
#pragma unroll
  for (int nt = 0; nt < 8; ++nt)
#pragma unroll
    for (int j = 0; j < 8; ++j) { const float v = acc[nt][j]; psum[j] += v; psq[j] += v * v; }
#pragma unroll
  for (int j = 0; j < 8; ++j) { psum[j] = red16_sum(psum[j]); psq[j] = red16_sum(psq[j]); }
  if (m == 0) {
#pragma unroll
    for (int j = 0; j < 8; ++j) {
      wstats[w][hh * 8 + j][0] = psum[j];
      wstats[w][hh * 8 + j][1] = psq[j];
    }
  }
  __syncthreads();
  if (tid < 16) {
    float s = 0.f, s2 = 0.f;
#pragma unroll
    for (int ww = 0; ww < 8; ++ww) { s += wstats[ww][tid][0]; s2 += wstats[ww][tid][1]; }
    const float mu  = s  * (1.f / 1024.f);
    const float var = s2 * (1.f / 1024.f) - mu * mu;
    fstat[tid][0] = mu;
    fstat[tid][1] = rsqrtf(var + 1e-5f);
  }
  __syncthreads();

  float mu[8], rs[8];
#pragma unroll
  for (int j = 0; j < 8; ++j) { mu[j] = fstat[hh * 8 + j][0]; rs[j] = fstat[hh * 8 + j][1]; }

#pragma unroll
  for (int nt = 0; nt < 8; ++nt) {
    const int col  = w * 128 + nt * 16 + m;
    const float gn = gain[col], bn = bias[col];
    const int head = col >> 6, d = col & 63;
#pragma unroll
    for (int j = 0; j < 8; ++j) {
      const int grow = rowbase + hh * 8 + j;
      const int bidx = grow >> 11;      // / SEQ
      const int sidx = grow & 2047;     // % SEQ
      const float v = (acc[nt][j] - mu[j]) * rs[j] * gn + bn;
      size_t o;
      if (transpose_out)
        o = ((size_t)(bidx * NHEAD + head) * DHEAD + d) * SEQ + sidx;   // [bh][d][s]
      else
        o = ((size_t)(bidx * NHEAD + head) * SEQ + sidx) * DHEAD + d;   // [bh][s][d]
      out[o] = (__bf16)v;
    }
  }
}

// ---------------------------------------------------------------------------
// Flash attention: one block = 128 query rows (8 waves x 16), one (b,h).
// K/V 64-key tiles staged once per block into double-buffered LDS via the
// async global->LDS path (overlaps DMA of block kb+1 with WMMA of block kb).
// Online softmax; all matmuls via WMMA bf16 / f32 accum.
// ---------------------------------------------------------------------------
__global__ __launch_bounds__(256) void k_attention(
    const __bf16* __restrict__ Q, const __bf16* __restrict__ K,
    const __bf16* __restrict__ Vt, const unsigned char* __restrict__ mask,
    __bf16* __restrict__ ctx)
{
  __shared__ __bf16 Ks[2][64][72];      // [key][d], padded
  __shared__ __bf16 Vs[2][64][72];      // [d][key], padded
  __shared__ __bf16 probs[8][16][72];   // per-wave prob tile, padded

  const int tid  = threadIdx.x;
  const int w    = tid >> 5;
  const int lane = tid & 31;
  const int hh   = lane >> 4;
  const int m    = lane & 15;
  const int bh   = blockIdx.y;
  const int bidx = bh >> 4, hidx = bh & 15;
  const int qbase = blockIdx.x * 128 + w * 16;

  const __bf16* Qh = Q  + (size_t)bh * SEQ * DHEAD;   // [s][d]
  const __bf16* Kh = K  + (size_t)bh * SEQ * DHEAD;   // [s][d]
  const __bf16* Vh = Vt + (size_t)bh * DHEAD * SEQ;   // [d][s]
  const unsigned char* Mb = mask + (size_t)bidx * SEQ * SEQ;

  // Stage one 64-key K and V tile (8 KB each) cooperatively: 4 x b128 per thread.
  auto stage_kv = [&](int keybase, int buf) {
#pragma unroll
    for (int i = 0; i < 2; ++i) {
      const int qidx = tid + i * 256;           // 0..511
      const int row  = qidx >> 3;               // 64 rows
      const int c8   = (qidx & 7) * 8;          // 8 chunks of 8 bf16 per row
      copy16_g2l(Kh + (size_t)(keybase + row) * DHEAD + c8, &Ks[buf][row][c8]);
      copy16_g2l(Vh + (size_t)row * SEQ + keybase + c8,     &Vs[buf][row][c8]);
    }
  };

  v16bf qfrag[2];
#pragma unroll
  for (int c = 0; c < 2; ++c) {
    const __bf16* p = Qh + (size_t)(qbase + m) * DHEAD + c * 32 + hh * 16;
    qfrag[c] = frag16(p, p + 8);
  }

  v8f acc[4] = {};
  float rm[8], rl[8];
#pragma unroll
  for (int j = 0; j < 8; ++j) { rm[j] = -3.0e38f; rl[j] = 0.f; }

  stage_kv(0, 0);

  const int NKB = SEQ / 64;
  for (int kb = 0; kb < NKB; ++kb) {
    const int keybase = kb * 64;
    const int buf = kb & 1;

    wait_async_done();     // this wave's staged tile landed in LDS
    __syncthreads();       // everyone's portion landed; prev-buf readers done
    if (kb + 1 < NKB) stage_kv(keybase + 64, buf ^ 1);   // overlap next DMA

    v8f sim[4];
#pragma unroll
    for (int nt = 0; nt < 4; ++nt) {
      v8f s = {};
#pragma unroll
      for (int c = 0; c < 2; ++c) {
        const __bf16* kp = &Ks[buf][nt * 16 + m][c * 32 + hh * 16];
        v16bf bf = frag16(kp, kp + 8);
        s = __builtin_amdgcn_wmma_f32_16x16x32_bf16(
            false, qfrag[c], false, bf, (short)0, s, false, false);
      }
#pragma unroll
      for (int j = 0; j < 8; ++j) {
        const int qrow = qbase + hh * 8 + j;
        const unsigned char mv = Mb[(size_t)qrow * SEQ + keybase + nt * 16 + m];
        s[j] = s[j] * 0.125f + (mv ? 0.f : -1e9f);   // scale 1/sqrt(64) + mask
      }
      sim[nt] = s;
    }
    // Row max across this 64-key block.
    float bm[8];
#pragma unroll
    for (int j = 0; j < 8; ++j) {
      float v = fmaxf(fmaxf(sim[0][j], sim[1][j]), fmaxf(sim[2][j], sim[3][j]));
      bm[j] = red16_max(v);
    }
    float nm[8], sc[8];
#pragma unroll
    for (int j = 0; j < 8; ++j) {
      nm[j] = fmaxf(rm[j], bm[j]);
      sc[j] = __expf(rm[j] - nm[j]);
      rm[j] = nm[j];
    }
    float bl[8] = {};
#pragma unroll
    for (int nt = 0; nt < 4; ++nt)
#pragma unroll
      for (int j = 0; j < 8; ++j) {
        const float p = __expf(sim[nt][j] - nm[j]);
        bl[j] += p;
        probs[w][hh * 8 + j][nt * 16 + m] = (__bf16)p;  // C-layout -> LDS
      }
#pragma unroll
    for (int j = 0; j < 8; ++j) {
      bl[j] = red16_sum(bl[j]);
      rl[j] = rl[j] * sc[j] + bl[j];
    }
#pragma unroll
    for (int t = 0; t < 4; ++t)
#pragma unroll
      for (int j = 0; j < 8; ++j) acc[t][j] *= sc[j];

    // P (A fragments from LDS) x V (B fragments from staged V^T tile).
#pragma unroll
    for (int c = 0; c < 2; ++c) {
      v16bf pa = frag16(&probs[w][m][c * 32 + hh * 16],
                        &probs[w][m][c * 32 + hh * 16 + 8]);
#pragma unroll
      for (int dt = 0; dt < 4; ++dt) {
        const __bf16* vp = &Vs[buf][dt * 16 + m][c * 32 + hh * 16];
        v16bf vb = frag16(vp, vp + 8);
        acc[dt] = __builtin_amdgcn_wmma_f32_16x16x32_bf16(
            false, pa, false, vb, (short)0, acc[dt], false, false);
      }
    }
  }

  float inv[8];
#pragma unroll
  for (int j = 0; j < 8; ++j) inv[j] = 1.f / rl[j];
#pragma unroll
  for (int dt = 0; dt < 4; ++dt)
#pragma unroll
    for (int j = 0; j < 8; ++j) {
      const int qrow = qbase + hh * 8 + j;
      ctx[(size_t)(bidx * SEQ + qrow) * DMODEL + hidx * DHEAD + dt * 16 + m] =
          (__bf16)(acc[dt][j] * inv[j]);
    }
}

// ---------------------------------------------------------------------------
// Output projection: ctx [8192,1024] bf16 @ Wot [n][k] bf16 + bo -> f32 out.
// ---------------------------------------------------------------------------
__global__ __launch_bounds__(256) void k_out_proj(
    const __bf16* __restrict__ ctx, const __bf16* __restrict__ Wt,
    const float* __restrict__ bo, float* __restrict__ out)
{
  __shared__ __bf16 At[16][1032];
  const int tid  = threadIdx.x;
  const int w    = tid >> 5;
  const int lane = tid & 31;
  const int hh   = lane >> 4;
  const int m    = lane & 15;
  const int rowbase = blockIdx.x * 16;

#pragma unroll 2
  for (int p = tid; p < 16 * 128; p += 256) {
    const int r = p >> 7, c8 = (p & 127) * 8;
    *(v8bf*)&At[r][c8] = *(const v8bf*)(ctx + (size_t)(rowbase + r) * DMODEL + c8);
  }
  __syncthreads();

  v8f acc[8] = {};
  for (int kk = 0; kk < 32; ++kk) {
    const int kb = kk * 32;
    v16bf afrag = frag16(&At[m][kb + hh * 16], &At[m][kb + hh * 16 + 8]);
#pragma unroll
    for (int nt = 0; nt < 8; ++nt) {
      const __bf16* wp = Wt + (size_t)(w * 128 + nt * 16 + m) * 1024 + kb + hh * 16;
      __builtin_prefetch(wp + 32, 0, 1);
      v16bf bfrag = frag16(wp, wp + 8);
      acc[nt] = __builtin_amdgcn_wmma_f32_16x16x32_bf16(
          false, afrag, false, bfrag, (short)0, acc[nt], false, false);
    }
  }

#pragma unroll
  for (int nt = 0; nt < 8; ++nt) {
    const int col  = w * 128 + nt * 16 + m;
    const float bn = bo[col];
#pragma unroll
    for (int j = 0; j < 8; ++j)
      out[(size_t)(rowbase + hh * 8 + j) * DMODEL + col] = acc[nt][j] + bn;
  }
}

// ---------------------------------------------------------------------------
extern "C" void kernel_launch(void* const* d_in, const int* in_sizes, int n_in,
                              void* d_out, int out_size, void* d_ws, size_t ws_size,
                              hipStream_t stream) {
  const float* q    = (const float*)d_in[0];
  const float* kin  = (const float*)d_in[1];
  const float* vin  = (const float*)d_in[2];
  const unsigned char* mask = (const unsigned char*)d_in[3];
  const float* Wq = (const float*)d_in[4];
  const float* gq = (const float*)d_in[5];
  const float* bq = (const float*)d_in[6];
  const float* Wk = (const float*)d_in[7];
  const float* gk = (const float*)d_in[8];
  const float* bk = (const float*)d_in[9];
  const float* Wv = (const float*)d_in[10];
  const float* gv = (const float*)d_in[11];
  const float* bv = (const float*)d_in[12];
  const float* Wo = (const float*)d_in[13];
  const float* bo = (const float*)d_in[14];
  float* out = (float*)d_out;

  char* ws = (char*)d_ws;
  const size_t MB = 1u << 20;
  __bf16* Wqt = (__bf16*)(ws + 0 * MB);     // 2 MB each
  __bf16* Wkt = (__bf16*)(ws + 2 * MB);
  __bf16* Wvt = (__bf16*)(ws + 4 * MB);
  __bf16* Wot = (__bf16*)(ws + 6 * MB);
  __bf16* Qb  = (__bf16*)(ws + 8 * MB);     // [bh][s][d]  16 MB
  __bf16* Kb  = (__bf16*)(ws + 24 * MB);    // [bh][s][d]  16 MB
  __bf16* Vtb = (__bf16*)(ws + 40 * MB);    // [bh][d][s]  16 MB
  __bf16* Ctx = (__bf16*)(ws + 56 * MB);    // [b*s][D]    16 MB

  // Weight convert + transpose to bf16.
  k_wconv<<<4096, 256, 0, stream>>>(Wq, Wqt);
  k_wconv<<<4096, 256, 0, stream>>>(Wk, Wkt);
  k_wconv<<<4096, 256, 0, stream>>>(Wv, Wvt);
  k_wconv<<<4096, 256, 0, stream>>>(Wo, Wot);

  // QKV projections + LayerNorm (V stored transposed for PV fragments).
  k_gemm_ln<<<512, 256, 0, stream>>>(q,   Wqt, gq, bq, Qb,  0);
  k_gemm_ln<<<512, 256, 0, stream>>>(kin, Wkt, gk, bk, Kb,  0);
  k_gemm_ln<<<512, 256, 0, stream>>>(vin, Wvt, gv, bv, Vtb, 1);

  // Attention: grid (S/128 q-tiles, B*H).
  k_attention<<<dim3(SEQ / 128, 4 * NHEAD), 256, 0, stream>>>(Qb, Kb, Vtb, mask, Ctx);

  // Output projection.
  k_out_proj<<<512, 256, 0, stream>>>(Ctx, Wot, bo, out);
}
